// GraphTransformerMultiHeadAttentionLayer_9981503996060
// MI455X (gfx1250) — matrix operand order repaired and checked by reading, not compiled, not run on hardware
//
#include <hip/hip_runtime.h>
#include <hip/hip_bf16.h>

// ---------------------------------------------------------------------------
// GraphTransformerMultiHeadAttentionLayer for MI455X (gfx1250, wave32, WMMA)
// B=8, N=1024, D=512, H=8, hd=64
// ---------------------------------------------------------------------------

#define B_  8
#define N_  1024
#define D_  512
#define H_  8
#define HD_ 64

typedef _Float16 half_t;
typedef __attribute__((ext_vector_type(16))) _Float16 v16h;
typedef __attribute__((ext_vector_type(8)))  _Float16 v8h;
typedef __attribute__((ext_vector_type(8)))  float    v8f;
typedef __attribute__((ext_vector_type(4)))  unsigned int v4u;
typedef __attribute__((ext_vector_type(4)))  int v4i;
typedef __attribute__((ext_vector_type(8)))  int v8i;

// ------------------------- WMMA fragment loaders ---------------------------
// A-matrix 16x32 f16, row-major source (leading dim ld, in halves).
// ISA layout: lanes 0-15 -> M=lane, K in {0..7, 16..23}; lanes 16-31 ->
// M=lane-16, K in {8..15, 24..31}.  (CDNA5 ISA 7.12.2)
__device__ __forceinline__ v16h load_a_f16(const half_t* p, int ld, int k0) {
    const int lane = threadIdx.x & 31;
    const int m    = lane & 15;
    const int kb   = k0 + ((lane >> 4) << 3);        // 0 or 8
    const half_t* r0 = p + (size_t)m * ld;
    v8h lo = *(const v8h*)(r0 + kb);
    v8h hi = *(const v8h*)(r0 + kb + 16);
    v16h a;
#pragma unroll
    for (int t = 0; t < 8; ++t) { a[t] = lo[t]; a[8 + t] = hi[t]; }
    return a;
}

// A-matrix 16x32 from an fp32 row-major source, converting to f16.
__device__ __forceinline__ v16h load_a_f32(const float* __restrict__ p, int ld, int k0) {
    const int lane = threadIdx.x & 31;
    const int m    = lane & 15;
    const int kb   = k0 + ((lane >> 4) << 3);
    const float* r0 = p + (size_t)m * ld;
    v16h a;
#pragma unroll
    for (int t = 0; t < 8; ++t) {
        a[t]     = (_Float16)r0[kb + t];
        a[8 + t] = (_Float16)r0[kb + 16 + t];
    }
    return a;
}

// B-matrix 32x16 where B[k][n] = p[n*ld + k]  (p holds B^T rows).  Lane
// n=lane&15 holds column n; lanes 0-15 K in {0..7, 16..23}, lanes 16-31 K in {8..15, 24..31}.
__device__ __forceinline__ v16h load_b_rows(const half_t* p, int ld, int k0) {
    const int lane = threadIdx.x & 31;
    const int n    = lane & 15;
    const int kb   = k0 + ((lane >> 4) << 3);        // 0 or 8
    const half_t* r = p + (size_t)n * ld + kb;
    v8h lo = *(const v8h*)(r);
    v8h hi = *(const v8h*)(r + 16);
    v16h bm;
#pragma unroll
    for (int t = 0; t < 8; ++t) { bm[t] = lo[t]; bm[8 + t] = hi[t]; }
    return bm;
}

#define WMMA_F16(a, b, c) \
    __builtin_amdgcn_wmma_f32_16x16x32_f16(false, (a), false, (b), (short)0, (c), false, false)

// --------------------------- fp32 -> f16 convert ---------------------------
__global__ void cvt_kernel(const float* __restrict__ in, half_t* __restrict__ out, int n) {
    int i = blockIdx.x * blockDim.x + threadIdx.x;
    int stride = gridDim.x * blockDim.x;
    for (; i < n; i += stride) out[i] = (_Float16)in[i];
}

// ------------------- QKV projection: Y = X*W^T + b -------------------------
// Q,K outputs in [B,H,N,hd]; V output transposed to [B,H,hd,N] so the AV
// GEMM can use contiguous K-major B-fragment loads.
__global__ void qkv_proj_kernel(const half_t* __restrict__ qh, const half_t* __restrict__ kh,
                                const half_t* __restrict__ vh,
                                const half_t* __restrict__ wq, const half_t* __restrict__ wk,
                                const half_t* __restrict__ wv,
                                const float* __restrict__ bq, const float* __restrict__ bk,
                                const float* __restrict__ bv,
                                half_t* __restrict__ Qp, half_t* __restrict__ Kp,
                                half_t* __restrict__ Vt) {
    const half_t* X; const half_t* W; const float* bias; half_t* Y;
    if (blockIdx.y == 0)      { X = qh; W = wq; bias = bq; Y = Qp; }
    else if (blockIdx.y == 1) { X = kh; W = wk; bias = bk; Y = Kp; }
    else                      { X = vh; W = wv; bias = bv; Y = Vt; }
    const bool vtrans = (blockIdx.y == 2);

    const int wid = blockIdx.x * (blockDim.x >> 5) + (threadIdx.x >> 5);
    const int dt = wid & 31;             // D/16 = 32 output-col tiles
    const int it = (wid >> 5) & 63;      // N/16 = 64 row tiles
    const int b  = wid >> 11;            // batch

    const half_t* xrow  = X + ((size_t)b * N_ + (size_t)it * 16) * D_;
    const half_t* wbase = W + (size_t)dt * 16 * D_;

    v8f acc = {};
#pragma unroll 4
    for (int k0 = 0; k0 < D_; k0 += 32) {
        v16h a  = load_a_f16(xrow, D_, k0);
        v16h bm = load_b_rows(wbase, D_, k0);
        acc = WMMA_F16(a, bm, acc);
    }

    const int lane = threadIdx.x & 31;
    const int n    = lane & 15;
    const int mofs = (lane >> 4) << 3;
    const int dout = dt * 16 + n;
    const int h    = dout >> 6;          // / hd
    const int dsub = dout & (HD_ - 1);
    const float bb = bias[dout];
#pragma unroll
    for (int r = 0; r < 8; ++r) {
        const int i = it * 16 + r + mofs;
        const _Float16 val = (_Float16)(acc[r] + bb);
        if (vtrans)
            Y[(((size_t)b * H_ + h) * HD_ + dsub) * N_ + i] = val;   // [B,H,hd,N]
        else
            Y[(((size_t)b * H_ + h) * N_ + i) * HD_ + dsub] = val;   // [B,H,N,hd]
    }
}

// ---------------- energy = fused(Q K^T, feature bias, masks) ---------------
// 256-thread block computes a 64x64 macro-tile.  Q block staged to LDS by
// cooperative b128 copies; K block staged by the Tensor Data Mover (TDM).
__global__ void energy_kernel(const half_t* __restrict__ Qp, const half_t* __restrict__ Kp,
                              const float* __restrict__ speaker, const int* __restrict__ dist,
                              const int* __restrict__ topic, const int* __restrict__ red,
                              const int* __restrict__ maskp,
                              const float* __restrict__ feat_w, const float* __restrict__ feat_b,
                              const float* __restrict__ comb_w, const float* __restrict__ comb_b,
                              const float* __restrict__ dist_tab, const float* __restrict__ topic_tab,
                              const float* __restrict__ red_tab,
                              float* __restrict__ attn) {
    __shared__ half_t sQ[64 * HD_];
    __shared__ half_t sK[64 * HD_];

    const int blk = blockIdx.x;               // B*H*16*16 = 16384 blocks
    const int tj0 = (blk & 15) * 64;
    const int ti0 = ((blk >> 4) & 15) * 64;
    const int h   = (blk >> 8) & 7;
    const int b   = blk >> 11;
    const size_t bh = (size_t)b * H_ + h;

    const half_t* Qblk = Qp + (bh * N_ + ti0) * HD_;   // 64 rows, fully contiguous
    const half_t* Kblk = Kp + (bh * N_ + tj0) * HD_;

    // Stage Q block (8 KB) cooperatively.
    {
        const v8h* src = (const v8h*)Qblk;
        v8h* dst = (v8h*)sQ;
        dst[threadIdx.x]       = src[threadIdx.x];
        dst[threadIdx.x + 256] = src[threadIdx.x + 256];
    }

    // Stage K block (8 KB) via TDM: 2D descriptor, 64x64 2-byte elements.
    if ((threadIdx.x >> 5) == 0) {
        const unsigned long long ga = (unsigned long long)(size_t)Kblk;
        const unsigned int lds_base = (unsigned int)(size_t)(&sK[0]);
        v4u g0;
        g0[0] = 1u;                                            // count=1 (valid D#)
        g0[1] = lds_base;                                      // lds_addr
        g0[2] = (unsigned int)ga;                              // global_addr[31:0]
        g0[3] = (unsigned int)((ga >> 32) & 0x1FFFFFFull)      // global_addr[56:32]
              | (2u << 30);                                    // type=2 ("image")
        v8i g1;
        g1[0] = (1 << 16);        // workgroup_mask=0, data_size=1 (2 bytes)
        g1[1] = (64 << 16);       // tensor_dim0 = 64
        g1[2] = (64 << 16);       // tensor_dim1 = 64
        g1[3] = (64 << 16);       // tile_dim0   = 64
        g1[4] = 64;               // tile_dim1   = 64
        g1[5] = 64;               // tensor_dim0_stride = 64 elements
        g1[6] = 0;
        g1[7] = 0;
        v4i gz = {0, 0, 0, 0};
#if __clang_major__ >= 23
        v8i gz8 = {0, 0, 0, 0, 0, 0, 0, 0};
        __builtin_amdgcn_tensor_load_to_lds(g0, g1, gz, gz, gz8, 0);
#else
        __builtin_amdgcn_tensor_load_to_lds(g0, g1, gz, gz, 0);
#endif
        __builtin_amdgcn_s_wait_tensorcnt(0);
    }
    __syncthreads();

    const int lane = threadIdx.x & 31;
    const int n    = lane & 15;
    const int mofs = (lane >> 4) << 3;
    const int wv   = threadIdx.x >> 5;

    const float fw0 = feat_w[h * 4 + 0], fw1 = feat_w[h * 4 + 1];
    const float fw2 = feat_w[h * 4 + 2], fw3 = feat_w[h * 4 + 3];
    const float fb  = feat_b[h];
    const float cw0 = comb_w[0], cw1 = comb_w[1], cb = comb_b[0];

    // Prefetch next macro-tile of the gather-heavy adjacency stream.
    __builtin_prefetch(dist + (((size_t)b * N_ + ti0) * N_ + tj0 + 64), 0, 0);

#pragma unroll
    for (int s = 0; s < 2; ++s) {
        const int t = wv + s * 8;                  // 16 sub-tiles over 8 waves
        const int tis = (t >> 2) & 3;
        const int tjs = t & 3;
        const half_t* qrow = sQ + (size_t)tis * 16 * HD_;
        const half_t* krow = sK + (size_t)tjs * 16 * HD_;

        v8f acc = {};
#pragma unroll
        for (int k0 = 0; k0 < HD_; k0 += 32) {
            v16h a  = load_a_f16(qrow, HD_, k0);
            v16h bm = load_b_rows(krow, HD_, k0);
            acc = WMMA_F16(a, bm, acc);
        }

        const int j  = tj0 + tjs * 16 + n;
        const int mv = maskp[(size_t)b * N_ + j];
#pragma unroll
        for (int r = 0; r < 8; ++r) {
            const int i = ti0 + tis * 16 + r + mofs;
            const size_t idx = ((size_t)b * N_ + i) * N_ + j;
            const int   rd  = red[idx];
            const float dv  = dist_tab[dist[idx]];
            const float sp  = speaker[idx];
            const float tv  = topic_tab[topic[idx]];
            const float rv  = red_tab[rd];
            float fco = tanhf(fw0 * dv + fw1 * sp + fw2 * tv + fw3 * rv + fb);
            float e   = acc[r] * 0.125f;            // 1/sqrt(hd), hd=64
            if (rd == 0) { fco = -10.0f; e = -15.0f; }
            float e2 = tanhf(cw0 * e + cw1 * fco + cb);
            if (mv == 0) e2 = -__builtin_inff();
            attn[(bh * N_ + i) * N_ + j] = e2;
        }
    }
}

// -------------------------- row softmax in place ---------------------------
__global__ void softmax_kernel(float* __restrict__ attn) {
    __shared__ float sdata[256];
    const size_t row = blockIdx.x;                  // B*H*N rows
    float* rowp = attn + row * (size_t)N_;
    const int tid = threadIdx.x;

    float v[4];
    float m = -__builtin_inff();
#pragma unroll
    for (int t = 0; t < 4; ++t) { v[t] = rowp[tid + 256 * t]; m = fmaxf(m, v[t]); }
    sdata[tid] = m; __syncthreads();
    for (int s = 128; s > 0; s >>= 1) {
        if (tid < s) sdata[tid] = fmaxf(sdata[tid], sdata[tid + s]);
        __syncthreads();
    }
    m = sdata[0]; __syncthreads();

    float lsum = 0.f;
#pragma unroll
    for (int t = 0; t < 4; ++t) { v[t] = __expf(v[t] - m); lsum += v[t]; }
    sdata[tid] = lsum; __syncthreads();
    for (int s = 128; s > 0; s >>= 1) {
        if (tid < s) sdata[tid] += sdata[tid + s];
        __syncthreads();
    }
    const float rinv = 1.0f / sdata[0];
#pragma unroll
    for (int t = 0; t < 4; ++t) rowp[tid + 256 * t] = v[t] * rinv;
}

// ------------------------- x = attention @ V -------------------------------
// V is pre-transposed to [B,H,hd,N]: B-fragment loads are contiguous b128s.
__global__ void av_kernel(const float* __restrict__ attn, const half_t* __restrict__ Vt,
                          half_t* __restrict__ Xh) {
    const int wid = blockIdx.x * (blockDim.x >> 5) + (threadIdx.x >> 5);
    const int dt = wid & 3;               // hd/16 = 4
    const int ti = (wid >> 2) & 63;
    const int h  = (wid >> 8) & 7;
    const int b  = wid >> 11;

    const size_t bh = (size_t)b * H_ + h;
    const float*  Prow = attn + (bh * N_ + (size_t)ti * 16) * N_;
    const half_t* Vb   = Vt + (bh * HD_ + (size_t)dt * 16) * N_;

    v8f acc = {};
    for (int j0 = 0; j0 < N_; j0 += 32) {
        v16h a  = load_a_f32(Prow, N_, j0);
        v16h bm = load_b_rows(Vb, N_, j0);
        acc = WMMA_F16(a, bm, acc);
    }

    const int lane = threadIdx.x & 31;
    const int n    = lane & 15;
    const int mofs = (lane >> 4) << 3;
    const int dout = h * HD_ + dt * 16 + n;
#pragma unroll
    for (int r = 0; r < 8; ++r) {
        const int i = ti * 16 + r + mofs;
        Xh[((size_t)b * N_ + i) * D_ + dout] = (_Float16)acc[r];
    }
}

// ----------------------- out = Xh @ Wo^T + bo (fp32) -----------------------
__global__ void out_proj_kernel(const half_t* __restrict__ Xh, const half_t* __restrict__ Wo,
                                const float* __restrict__ bo, float* __restrict__ out) {
    const int wid = blockIdx.x * (blockDim.x >> 5) + (threadIdx.x >> 5);
    const int dt = wid & 31;
    const int it = (wid >> 5) & 63;
    const int b  = wid >> 11;

    const half_t* xrow  = Xh + ((size_t)b * N_ + (size_t)it * 16) * D_;
    const half_t* wbase = Wo + (size_t)dt * 16 * D_;

    v8f acc = {};
#pragma unroll 4
    for (int k0 = 0; k0 < D_; k0 += 32) {
        v16h a  = load_a_f16(xrow, D_, k0);
        v16h bm = load_b_rows(wbase, D_, k0);
        acc = WMMA_F16(a, bm, acc);
    }

    const int lane = threadIdx.x & 31;
    const int n    = lane & 15;
    const int mofs = (lane >> 4) << 3;
    const int dout = dt * 16 + n;
    const float bb = bo[dout];
#pragma unroll
    for (int r = 0; r < 8; ++r) {
        const int i = it * 16 + r + mofs;
        out[((size_t)b * N_ + i) * D_ + dout] = acc[r] + bb;
    }
}

// ---------------------------------------------------------------------------
extern "C" void kernel_launch(void* const* d_in, const int* in_sizes, int n_in,
                              void* d_out, int out_size, void* d_ws, size_t ws_size,
                              hipStream_t stream) {
    (void)in_sizes; (void)n_in; (void)out_size; (void)ws_size;

    const float* query   = (const float*)d_in[0];
    const float* key     = (const float*)d_in[1];
    const float* value   = (const float*)d_in[2];
    const float* speaker = (const float*)d_in[3];
    const int*   dist    = (const int*)d_in[4];
    const int*   topic   = (const int*)d_in[5];
    const int*   red     = (const int*)d_in[6];
    const int*   maskp   = (const int*)d_in[7];
    const float* Wq      = (const float*)d_in[8];
    const float* bq      = (const float*)d_in[9];
    const float* Wk      = (const float*)d_in[10];
    const float* bk      = (const float*)d_in[11];
    const float* Wv      = (const float*)d_in[12];
    const float* bv      = (const float*)d_in[13];
    const float* Wo      = (const float*)d_in[14];
    const float* bo      = (const float*)d_in[15];
    const float* feat_w  = (const float*)d_in[16];
    const float* feat_b  = (const float*)d_in[17];
    const float* comb_w  = (const float*)d_in[18];
    const float* comb_b  = (const float*)d_in[19];
    const float* dist_tab  = (const float*)d_in[20];
    const float* topic_tab = (const float*)d_in[21];
    const float* red_tab   = (const float*)d_in[22];

    const size_t BND = (size_t)B_ * N_ * D_;     // 4,194,304
    const size_t DD  = (size_t)D_ * D_;          // 262,144

    float* out_x = (float*)d_out;
    float* attn  = out_x + BND;                  // [B,H,N,N]

    half_t* ws  = (half_t*)d_ws;
    half_t* qh  = ws;
    half_t* kh  = qh + BND;
    half_t* vh  = kh + BND;
    half_t* wqh = vh + BND;
    half_t* wkh = wqh + DD;
    half_t* wvh = wkh + DD;
    half_t* woh = wvh + DD;
    half_t* Qp  = woh + DD;
    half_t* Kp  = Qp + BND;
    half_t* Vt  = Kp + BND;                      // [B,H,hd,N]
    half_t* Xh  = Vt + BND;

    // 1) convert inputs / weights to f16
    cvt_kernel<<<2048, 256, 0, stream>>>(query, qh, (int)BND);
    cvt_kernel<<<2048, 256, 0, stream>>>(key,   kh, (int)BND);
    cvt_kernel<<<2048, 256, 0, stream>>>(value, vh, (int)BND);
    cvt_kernel<<<256,  256, 0, stream>>>(Wq, wqh, (int)DD);
    cvt_kernel<<<256,  256, 0, stream>>>(Wk, wkh, (int)DD);
    cvt_kernel<<<256,  256, 0, stream>>>(Wv, wvh, (int)DD);
    cvt_kernel<<<256,  256, 0, stream>>>(Wo, woh, (int)DD);

    // 2) QKV projections (16384 waves each, 8 waves / 256-thread block)
    qkv_proj_kernel<<<dim3(2048, 3), 256, 0, stream>>>(qh, kh, vh, wqh, wkh, wvh,
                                                       bq, bk, bv, Qp, Kp, Vt);

    // 3) fused energy + feature bias (16384 blocks, 64x64 macro-tiles, TDM+LDS)
    energy_kernel<<<16384, 256, 0, stream>>>(Qp, Kp, speaker, dist, topic, red, maskp,
                                             feat_w, feat_b, comb_w, comb_b,
                                             dist_tab, topic_tab, red_tab, attn);

    // 4) softmax rows in place (B*H*N = 65536 rows)
    softmax_kernel<<<65536, 256, 0, stream>>>(attn);

    // 5) x = attn @ V   (16384 waves)
    av_kernel<<<2048, 256, 0, stream>>>(attn, Vt, Xh);

    // 6) output projection (16384 waves)
    out_proj_kernel<<<2048, 256, 0, stream>>>(Xh, woh, bo, out_x);
}
